// GeometricSparseNeighConsensus_35416300322858
// MI455X (gfx1250) — matrix operand order repaired and checked by reading, not compiled, not run on hardware
//
#include <hip/hip_runtime.h>

typedef __attribute__((ext_vector_type(16))) _Float16 v16h;
typedef __attribute__((ext_vector_type(8)))  float    v8f;

#define DD 32
#define TH1 2
#define TW1 2
#define TH2 4
#define TW2 16
// staged x tile dims (with +-1 halo in the 4 spatial dims), 4 scale slices
#define XS1 (TH1+2)
#define XS2 (TW1+2)
#define XS3 (TH2+2)
#define XS4 (TW2+2)
#define XS_TOTAL (4*XS1*XS2*XS3*XS4)   // 6912 == 27*256
#define NCOL (4*TH1*TW1*TH2*TW2)       // 1024
#define VW 20                          // 18 used channels + 2 dead pad slots

// LDS offset (in halves) of tap t relative to &xs[s][ih1][iw1][ih2][n].
// t = dh1*27 + dw1*9 + dh2*3 + dw2. Taps t>=81 are zero-weighted in A, so
// clamp them to a valid address (value is a don't-care).
__device__ constexpr int tap_delta(int t) {
    const int tt  = t < 81 ? t : 80;
    const int dw2 = tt % 3, dh2 = (tt/3) % 3, dw1 = (tt/9) % 3, dh1 = tt/27;
    return dh1*(XS2*XS3*XS4) + dw1*(XS3*XS4) + dh2*XS4 + dw2;
}

__global__ __launch_bounds__(256) void conv6d_wmma_kernel(
    const float* __restrict__ x, const float* __restrict__ kern,
    float* __restrict__ out)
{
    __shared__ _Float16 xs[4][XS1][XS2][XS3][XS4];   // 13824 B
    __shared__ _Float16 kmat[32][96];                //  6144 B
    __shared__ _Float16 vbuf[NCOL][VW];              // 40960 B

    const int tid  = threadIdx.x;
    const int lane = tid & 31;
    const int wid  = tid >> 5;
    const bool hi  = lane >= 16;

    // decode 1D block id -> tile origin
    int b = blockIdx.x;
    const int tw2 = b & 1;  b >>= 1;
    const int th2 = b & 7;  b >>= 3;
    const int tw1 = b & 15; b >>= 4;
    const int th1 = b;
    const int H0 = th1*TH1, W0 = tw1*TW1, HH0 = th2*TH2, WW0 = tw2*TW2;

    // ---------- Phase 0: stage weight matrix kmat[m][k] into LDS ----------
    // rows m: 0..8 -> kernel k (a = m), 9..17 -> permuted kernel kT (a = m-9),
    // 18..31 -> zero.  cols k: tap t (k<81), zero-padded to 96.
    for (int i = tid; i < 32*96; i += 256) {         // 12 iterations, uniform
        const int m = i / 96, k = i % 96;
        const int t   = k < 81 ? k : 80;
        const int dw2 = t % 3, dh2 = (t/3) % 3, dw1 = (t/9) % 3, dh1 = t/27;
        const int a0  = m < 9 ? m : m - 9;
        const int a   = a0 < 9 ? a0 : 8;             // clamp rows >= 18
        const int o1  = a / 3, o2 = a % 3;
        const int kidx0 = a*81 + t;                                        // plain k
        const int kidxT = o2*243 + o1*81 + dh2*27 + dw2*9 + dh1*3 + dw1;   // kT
        const float w = kern[(m >= 9) ? kidxT : kidx0];
        const bool valid = (k < 81) && (m < 18);
        kmat[m][k] = valid ? (_Float16)w : (_Float16)0.f;
    }

    // ---------- Phase 1: stage relu(x) tile (+halo, zero-padded) as f16 ----------
    for (int i = tid; i < XS_TOTAL; i += 256) {      // 27 iterations, uniform
        int r = i;
        const int i4 = r % XS4; r /= XS4;
        const int i3 = r % XS3; r /= XS3;
        const int i2 = r % XS2; r /= XS2;
        const int i1 = r % XS1; r /= XS1;
        const int s  = r;
        const int h1 = H0 + i1 - 1, w1 = W0 + i2 - 1;
        const int h2 = HH0 + i3 - 1, w2 = WW0 + i4 - 1;
        const bool inb = ((unsigned)h1 < DD) & ((unsigned)w1 < DD) &
                         ((unsigned)h2 < DD) & ((unsigned)w2 < DD);
        // wrap with &31: any in-bounds address is fine, value selected below
        const float xv = x[((((size_t)s*DD + (h1 & 31))*DD + (w1 & 31))*DD
                            + (h2 & 31))*DD + (w2 & 31)];
        const float v = (inb && xv > 0.f) ? xv : 0.f;
        (&xs[0][0][0][0][0])[i] = (_Float16)v;
    }

    __syncthreads();

    // ---------- Build A fragments from kmat (once per block) ----------
    // ISA 16-bit A(16x32) layout: lane<16: K = e<8 ? e : e+8 ; lane>=16: K = e<8 ? e+8 : e+16
    const int mrow = lane & 15;
    const _Float16* kb0 = &kmat[mrow][0];
    const _Float16* kb1 = &kmat[mrow + 16][0];
    v16h A0[3], A1[3];
    #pragma unroll
    for (int c = 0; c < 3; ++c) {
        #pragma unroll
        for (int e = 0; e < 16; ++e) {
            const int klo = 32*c + (e < 8 ? e     : e + 8);
            const int khi = 32*c + (e < 8 ? e + 8 : e + 16);
            const int ko  = hi ? khi : klo;          // select of two constants
            A0[c][e] = kb0[ko];
            A1[c][e] = kb1[ko];
        }
    }

    // ---------- Precompute B gather offsets (loop-invariant, select of consts) ----------
    int dlt[48];
    #pragma unroll
    for (int j = 0; j < 48; ++j) {
        const int c = j >> 4, e = j & 15;
        dlt[j] = hi ? tap_delta(32*c + (e < 8 ? e + 8 : e + 16)) : tap_delta(32*c + (e < 8 ? e : e + 8));
    }

    // ---------- Phase 2: v[m,col] = Kmat . u via WMMA ----------
    // B(32x16) layout: column N = lane&15, lane<16: K = e<8 ? e : e+8 ; lane>=16: K = e<8 ? e+8 : e+16
    const int n = lane & 15;
    const _Float16* xsf = &xs[0][0][0][0][0];
    for (int gi = 0; gi < 8; ++gi) {
        const int g   = wid*8 + gi;                  // 0..63 column groups
        const int ih2 = g & 3, iw1 = (g>>2)&1, ih1 = (g>>3)&1, s = g>>4;
        const _Float16* basep =
            xsf + (((s*XS1 + ih1)*XS2 + iw1)*XS3 + ih2)*XS4 + n;
        v8f acc0 = {}; v8f acc1 = {};
        #pragma unroll
        for (int c = 0; c < 3; ++c) {
            v16h B;
            #pragma unroll
            for (int e = 0; e < 16; ++e) B[e] = basep[dlt[c*16 + e]];
            acc0 = __builtin_amdgcn_wmma_f32_16x16x32_f16(
                       false, A0[c], false, B, (short)0, acc0, false, false);
            acc1 = __builtin_amdgcn_wmma_f32_16x16x32_f16(
                       false, A1[c], false, B, (short)0, acc1, false, false);
        }
        // D layout: lane holds column N = lane&15, rows M = e + (hi ? 8 : 0)
        const int col   = (g << 4) | n;
        const int mbase = hi ? 8 : 0;
        #pragma unroll
        for (int e = 0; e < 8; ++e) vbuf[col][mbase + e] = (_Float16)acc0[e];
        // tile1: rows 16,17 live in lanes<16 (e=0,1); hi lanes hold dead rows
        // 24,25 -> park them in pad slots 18,19 (never read) to stay branch-free
        const int m2 = hi ? 18 : 16;
        vbuf[col][m2]     = (_Float16)acc1[0];
        vbuf[col][m2 + 1] = (_Float16)acc1[1];
    }

    __syncthreads();

    // ---------- Phase 3: scale-dim stencil + sigmoid + mask + sum over scales ----------
    const int ih1 = tid >> 7, iw1 = (tid >> 6) & 1, ih2 = (tid >> 4) & 3, iw2 = tid & 15;
    const int h1g = H0+ih1, w1g = W0+iw1, h2g = HH0+ih2, w2g = WW0+iw2;
    float outv = 0.f;
    #pragma unroll
    for (int s1o = 0; s1o < 2; ++s1o) {
        #pragma unroll
        for (int s2o = 0; s2o < 2; ++s2o) {
            float z0 = 0.f, z1 = 0.f;
            #pragma unroll
            for (int o1 = 0; o1 < 3; ++o1) {
                const int s1p = s1o + o1 - 1;
                if (s1p < 0 || s1p > 1) continue;    // compile-time resolved
                #pragma unroll
                for (int o2 = 0; o2 < 3; ++o2) {
                    const int s2p = s2o + o2 - 1;
                    if (s2p < 0 || s2p > 1) continue;
                    const int col = ((((s1p*2+s2p)*TH1 + ih1)*TW1 + iw1)*TH2 + ih2)*TW2 + iw2;
                    const int a = o1*3 + o2;
                    z0 += (float)vbuf[col][a];
                    z1 += (float)vbuf[col][9 + a];
                }
            }
            const float xv = x[((((size_t)(s1o*2+s2o)*DD + h1g)*DD + w1g)*DD + h2g)*DD + w2g];
            const float sig = 1.f/(1.f + __expf(-z0)) + 1.f/(1.f + __expf(-z1));
            outv += (xv != 0.f) ? sig : 0.f;
        }
    }
    out[(((size_t)h1g*DD + w1g)*DD + h2g)*DD + w2g] = outv;
}

extern "C" void kernel_launch(void* const* d_in, const int* in_sizes, int n_in,
                              void* d_out, int out_size, void* d_ws, size_t ws_size,
                              hipStream_t stream) {
    (void)in_sizes; (void)n_in; (void)out_size; (void)d_ws; (void)ws_size;
    const float* x    = (const float*)d_in[0];   // (1,2,2,32,32,32,32) f32
    const float* kern = (const float*)d_in[1];   // (729,) f32
    float* out = (float*)d_out;                  // (1,32,32,32,32) f32
    // tiles: 16 (h1) * 16 (w1) * 8 (h2) * 2 (w2) = 4096 blocks of 256 threads
    conv6d_wmma_kernel<<<dim3(4096), dim3(256), 0, stream>>>(x, kern, out);
}
